// Model_60713657696372
// MI455X (gfx1250) — compile-verified, matrix operand order checked by reading, not run on hardware
//
#include <hip/hip_runtime.h>
#include <hip/hip_bf16.h>

typedef __attribute__((ext_vector_type(16))) _Float16 v16h;
typedef __attribute__((ext_vector_type(8)))  _Float16 v8h;
typedef __attribute__((ext_vector_type(8)))  float    v8f;
typedef int v4i_vs __attribute__((vector_size(16)));   // matches builtin param pointee

#define AS1 __attribute__((address_space(1)))
#define AS3 __attribute__((address_space(3)))

static constexpr int Bn = 2, Cc = 64;
static constexpr int HW = 128 * 128;        // 16384
static constexpr int NT = Bn * HW;          // 32768 tokens / pixels
static constexpr int XPF = 36;              // dt_rank(4) + 2*D_STATE(32)
static constexpr int CHUNK = 128;
static constexpr int NCHUNK = HW / CHUNK;   // 128

#if __has_builtin(__builtin_amdgcn_global_load_async_to_lds_b128) && \
    __has_builtin(__builtin_amdgcn_s_wait_asynccnt)
#define USE_ASYNC_LDS 1
#else
#define USE_ASYNC_LDS 0
#endif

static __device__ __forceinline__ float siluf(float v){ return v / (1.f + expf(-v)); }
static __device__ __forceinline__ float softplusf(float v){ return (v > 20.f) ? v : log1pf(expf(v)); }

// ---------------------------------------------------------------- weights->f16
__global__ __launch_bounds__(256) void k_wconv(const float* __restrict__ inw,
                                               const float* __restrict__ outw,
                                               _Float16* __restrict__ inw_h,
                                               _Float16* __restrict__ outw_h){
  int g = blockIdx.x * 256 + threadIdx.x;
  if (g < 256 * 64)  inw_h[g]  = (_Float16)inw[g];
  if (g < 64 * 128)  outw_h[g] = (_Float16)outw[g];
}

// ------------------------------------------------- x (b,c,hw) -> xs f16 (t,c)
__global__ __launch_bounds__(256) void k_transpose(const float* __restrict__ x,
                                                   _Float16* __restrict__ xs){
  __shared__ float tile[32][33];
  int hw0 = blockIdx.x * 32, c0 = blockIdx.y * 32, b = blockIdx.z;
  for (int i = threadIdx.y; i < 32; i += 8)
    tile[i][threadIdx.x] = x[(size_t)((b * 64 + c0 + i)) * HW + hw0 + threadIdx.x];
  __syncthreads();
  for (int i = threadIdx.y; i < 32; i += 8)
    xs[(size_t)(b * HW + hw0 + i) * 64 + c0 + threadIdx.x] = (_Float16)tile[threadIdx.x][i];
}

// ------------------------------------------------------------- WMMA f16 GEMM
// C(M,N) = A(M,K) * W(N,K)^T ; K multiple of 32, N multiple of 32, M multiple of 16.
// ISA 7.12.2 16-bit A 16x32 fragment: lane<16 holds K=[0..7],[16..23],
// lane>=16 holds K=[8..15],[24..31]  -> two 16B chunks at koff and koff+16 halves.
static __device__ __forceinline__ v16h load_frag16(const _Float16* __restrict__ base,
                                                   int row, int ld, int k0, int lane){
  const _Float16* p = base + (size_t)row * ld + k0 + ((lane >> 4) << 3);
  v8h lo = *(const v8h*)(p);
  v8h hi = *(const v8h*)(p + 16);
  v16h r;
#pragma unroll
  for (int i = 0; i < 8; ++i){ r[i] = lo[i]; r[i + 8] = hi[i]; }
  return r;
}

// One block handles a full 16-row M-slab across all of N.
// The 16xK A-slab is staged into LDS (async global->LDS on CDNA5), shared by all
// waves; each wave computes NTILE adjacent 16x16 tiles reusing one A fragment.
template<int N, int K, int NTILE>
__global__ __launch_bounds__(256) void k_gemm_wmma(const _Float16* __restrict__ A,
                                                   const _Float16* __restrict__ W,
                                                   float* __restrict__ C){
  constexpr int NTL  = N / 16;
  constexpr int WPB  = NTL / NTILE;       // waves per block
  constexpr int PADK = K + 16;            // 32B row pad: 16B-aligned, fewer bank conflicts
  constexpr int CPR  = K / 8;             // 16B chunks per A row
  constexpr int CH   = 16 * CPR;          // total chunks in the 16xK slab
  __shared__ __align__(16) _Float16 sA[16 * PADK];

  int tid  = threadIdx.x;
  int lane = tid & 31;
  int w    = tid >> 5;
  int tm   = blockIdx.x;

  const _Float16* Ab = A + (size_t)tm * 16 * K;
  for (int c = tid; c < CH; c += WPB * 32){
    int r = c / CPR, o = c - r * CPR;
#if USE_ASYNC_LDS
    __builtin_amdgcn_global_load_async_to_lds_b128(
        (AS1 v4i_vs*)(Ab + r * K + o * 8),
        (AS3 v4i_vs*)(&sA[r * PADK + o * 8]), 0, 0);
#else
    *(v8h*)(&sA[r * PADK + o * 8]) = *(const v8h*)(Ab + r * K + o * 8);
#endif
  }
#if USE_ASYNC_LDS
  __builtin_amdgcn_s_wait_asynccnt(0);
#endif
  __syncthreads();

  v8f acc[NTILE];
#pragma unroll
  for (int j = 0; j < NTILE; ++j) acc[j] = (v8f){};

  int arow = lane & 15;
#pragma unroll
  for (int k0 = 0; k0 < K; k0 += 32){
    v16h a = load_frag16(sA, arow, PADK, k0, lane);   // ds_load_b128 x2
#pragma unroll
    for (int j = 0; j < NTILE; ++j){
      int rW = ((w * NTILE + j) << 4) + (lane & 15);
      v16h b = load_frag16(W, rW, K, k0, lane);
      acc[j] = __builtin_amdgcn_wmma_f32_16x16x32_f16(false, a, false, b, (short)0, acc[j],
                                                      false, false);
    }
  }
  int row0 = (tm << 4) + ((lane >> 4) << 3);
#pragma unroll
  for (int j = 0; j < NTILE; ++j){
    int col = ((w * NTILE + j) << 4) + (lane & 15);
#pragma unroll
    for (int r = 0; r < 8; ++r) C[(size_t)(row0 + r) * N + col] = acc[j][r];
  }
}

// -------------------------------------------------- depthwise causal conv + silu
__global__ __launch_bounds__(256) void k_conv(const float* __restrict__ xz,
                                              const float* __restrict__ cw,
                                              const float* __restrict__ cb,
                                              float* __restrict__ xcv){
  int g = blockIdx.x * 256 + threadIdx.x;   // NT*128
  int t = g >> 7, ch = g & 127;
  int l = t & (HW - 1);
  float acc = cb[ch];
#pragma unroll
  for (int k = 0; k < 4; ++k){
    int ll = l - 3 + k;
    if (ll >= 0) acc += xz[(size_t)(t - l + ll) * 256 + ch] * cw[ch * 4 + k];
  }
  xcv[g] = siluf(acc);
}

// ------------------------------------------------------------------- x_proj
__global__ __launch_bounds__(256) void k_xproj(const float* __restrict__ xcv,
                                               const float* __restrict__ xpw,
                                               float* __restrict__ dbl){
  int g = blockIdx.x * 256 + threadIdx.x;   // NT*36
  if (g >= NT * XPF) return;
  int t = g / XPF, f = g - t * XPF;
  const float* xr = xcv + (size_t)t * 128;
  const float* wr = xpw + f * 128;
  float acc = 0.f;
#pragma unroll 4
  for (int ch = 0; ch < 128; ++ch) acc += xr[ch] * wr[ch];
  dbl[g] = acc;
}

// -------------------------------------------------------------------- delta
__global__ __launch_bounds__(256) void k_delta(const float* __restrict__ dbl,
                                               const float* __restrict__ dtw,
                                               const float* __restrict__ dtb,
                                               float* __restrict__ delta){
  int g = blockIdx.x * 256 + threadIdx.x;   // NT*128
  int t = g >> 7, ch = g & 127;
  const float* dr = dbl + (size_t)t * XPF;
  float acc = dtb[ch];
#pragma unroll
  for (int r = 0; r < 4; ++r) acc += dr[r] * dtw[ch * 4 + r];
  delta[g] = softplusf(acc);
}

// ------------------------------------------------ chunked scan, pass1: chunk aggregates
__global__ __launch_bounds__(256) void k_scan1(const float* __restrict__ delta,
                                               const float* __restrict__ xcv,
                                               const float* __restrict__ dbl,
                                               const float* __restrict__ A_log,
                                               float* __restrict__ cP, float* __restrict__ cS){
  int tid = blockIdx.x * 256 + threadIdx.x; // 2*128*16*NCHUNK lanes
  int s = tid & 15;
  int rest = tid >> 4;
  int chunk = rest & (NCHUNK - 1);
  int bc = rest >> 7;                       // b*128+ch
  int b = bc >> 7, ch = bc & 127;
  float Ac = -expf(A_log[ch * 16 + s]);
  float h = 0.f, P = 1.f;
  int lbase = chunk * CHUNK;
  for (int i = 0; i < CHUNK; ++i){
    int t = (b << 14) + lbase + i;
    float d  = delta[(size_t)t * 128 + ch];
    float u  = xcv[(size_t)t * 128 + ch];
    float Bv = dbl[(size_t)t * XPF + 4 + s];
    float dA = expf(d * Ac);
    h = h * dA + d * u * Bv;
    P *= dA;
  }
  cP[(size_t)(bc * 16 + s) * NCHUNK + chunk] = P;
  cS[(size_t)(bc * 16 + s) * NCHUNK + chunk] = h;
}

// ------------------------------------------------ pass2: prefix over chunks
__global__ __launch_bounds__(256) void k_scan2(const float* __restrict__ cP,
                                               const float* __restrict__ cS,
                                               float* __restrict__ cH){
  int idx = blockIdx.x * 256 + threadIdx.x; // 4096 = 256*16
  const float* Pp = cP + (size_t)idx * NCHUNK;
  const float* Sp = cS + (size_t)idx * NCHUNK;
  float* Hp = cH + (size_t)idx * NCHUNK;
  float h = 0.f;
  for (int c = 0; c < NCHUNK; ++c){
    Hp[c] = h;
    h = Pp[c] * h + Sp[c];
  }
}

// ------------------------------------------------ pass3: replay + y reduction
__global__ __launch_bounds__(256) void k_scan3(const float* __restrict__ delta,
                                               const float* __restrict__ xcv,
                                               const float* __restrict__ dbl,
                                               const float* __restrict__ A_log,
                                               const float* __restrict__ cH,
                                               float* __restrict__ yout){
  int tid = blockIdx.x * 256 + threadIdx.x;
  int s = tid & 15;
  int rest = tid >> 4;
  int chunk = rest & (NCHUNK - 1);
  int bc = rest >> 7;
  int b = bc >> 7, ch = bc & 127;
  float Ac = -expf(A_log[ch * 16 + s]);
  float h = cH[(size_t)(bc * 16 + s) * NCHUNK + chunk];
  int lbase = chunk * CHUNK;
  for (int i = 0; i < CHUNK; ++i){
    int t = (b << 14) + lbase + i;
    float d  = delta[(size_t)t * 128 + ch];
    float u  = xcv[(size_t)t * 128 + ch];
    float Bv = dbl[(size_t)t * XPF + 4 + s];
    float Cv = dbl[(size_t)t * XPF + 20 + s];
    h = h * expf(d * Ac) + d * u * Bv;
    float part = h * Cv;
    part += __shfl_xor(part, 8, 32);
    part += __shfl_xor(part, 4, 32);
    part += __shfl_xor(part, 2, 32);
    part += __shfl_xor(part, 1, 32);
    if (s == 0) yout[(size_t)t * 128 + ch] = part;
  }
}

// ---------------------------------------------- y2 = (y + xc*D)*silu(z) -> f16
__global__ __launch_bounds__(256) void k_ymul(const float* __restrict__ ysc,
                                              const float* __restrict__ xcv,
                                              const float* __restrict__ xz,
                                              const float* __restrict__ Dp,
                                              _Float16* __restrict__ y2h){
  int g = blockIdx.x * 256 + threadIdx.x;   // NT*128
  int t = g >> 7, ch = g & 127;
  float z = xz[(size_t)t * 256 + 128 + ch];
  float v = (ysc[g] + xcv[g] * Dp[ch]) * siluf(z);
  y2h[g] = (_Float16)v;
}

// -------------------------------------------- spectral branch, fused per-pixel
__global__ __launch_bounds__(256) void k_spe(const float* __restrict__ x,
                                             const float* __restrict__ in_w,
                                             const float* __restrict__ cw,
                                             const float* __restrict__ cb,
                                             const float* __restrict__ xpw,
                                             const float* __restrict__ dtw,
                                             const float* __restrict__ dtb,
                                             const float* __restrict__ A_log,
                                             const float* __restrict__ Dp,
                                             const float* __restrict__ ow,
                                             float* __restrict__ spey){
  __shared__ float s_x[8][64];
  __shared__ float s_xc[8][8][16];
  __shared__ float s_z[8][8][16];
  __shared__ float s_xcv[8][8][16];
  __shared__ float s_dbl[8][8][33];
  __shared__ float s_dl[8][8][16];
  __shared__ float s_y[8][8][16];
  int w = threadIdx.x >> 5;
  int lane = threadIdx.x & 31;
  int pix = blockIdx.x * 8 + w;             // b*HW + hw
  int b = pix >> 14, hw = pix & (HW - 1);
  for (int c = lane; c < 64; c += 32)
    s_x[w][c] = x[(size_t)(b * 64 + c) * HW + hw];
  __syncthreads();
  { // in_proj: e = lane (0..31 = 2*d_inner)
    int e = lane;
#pragma unroll
    for (int l = 0; l < 8; ++l){
      float acc = 0.f;
#pragma unroll
      for (int j = 0; j < 8; ++j) acc += s_x[w][l * 8 + j] * in_w[e * 8 + j];
      if (e < 16) s_xc[w][l][e] = acc; else s_z[w][l][e - 16] = acc;
    }
  }
  __syncthreads();
  { // causal depthwise conv + silu
    int ch = lane & 15, lh = lane >> 4;
#pragma unroll
    for (int l = lh * 4; l < lh * 4 + 4; ++l){
      float acc = cb[ch];
#pragma unroll
      for (int k = 0; k < 4; ++k){
        int ll = l - 3 + k;
        if (ll >= 0) acc += s_xc[w][ll][ch] * cw[ch * 4 + k];
      }
      s_xcv[w][l][ch] = siluf(acc);
    }
  }
  __syncthreads();
  for (int idx = lane; idx < 264; idx += 32){ // x_proj: 8 l * 33 f
    int l = idx / 33, f = idx - l * 33;
    float acc = 0.f;
#pragma unroll
    for (int ch = 0; ch < 16; ++ch) acc += s_xcv[w][l][ch] * xpw[f * 16 + ch];
    s_dbl[w][l][f] = acc;
  }
  __syncthreads();
  for (int idx = lane; idx < 128; idx += 32){ // delta
    int l = idx >> 4, ch = idx & 15;
    s_dl[w][l][ch] = softplusf(s_dbl[w][l][0] * dtw[ch] + dtb[ch]);
  }
  __syncthreads();
  { // scan over 8 tokens; lane = (ch, state-half of 8)
    int ch = lane & 15, sh = lane >> 4;
    float hs[8], Ac[8];
#pragma unroll
    for (int i = 0; i < 8; ++i){ hs[i] = 0.f; Ac[i] = -expf(A_log[ch * 16 + sh * 8 + i]); }
    for (int l = 0; l < 8; ++l){
      float d = s_dl[w][l][ch];
      float du = d * s_xcv[w][l][ch];
      float part = 0.f;
#pragma unroll
      for (int i = 0; i < 8; ++i){
        int s = sh * 8 + i;
        hs[i] = hs[i] * expf(d * Ac[i]) + du * s_dbl[w][l][1 + s];
        part += hs[i] * s_dbl[w][l][17 + s];
      }
      part += __shfl_xor(part, 16, 32);
      if (sh == 0) s_y[w][l][ch] = part;
    }
  }
  __syncthreads();
  for (int idx = lane; idx < 128; idx += 32){ // gate
    int l = idx >> 4, ch = idx & 15;
    s_y[w][l][ch] = (s_y[w][l][ch] + s_xcv[w][l][ch] * Dp[ch]) * siluf(s_z[w][l][ch]);
  }
  __syncthreads();
  for (int idx = lane; idx < 64; idx += 32){ // out_proj; c = l*8+d
    int l = idx >> 3, d = idx & 7;
    float acc = 0.f;
#pragma unroll
    for (int ch = 0; ch < 16; ++ch) acc += s_y[w][l][ch] * ow[d * 16 + ch];
    spey[(size_t)pix * 64 + l * 8 + d] = acc;
  }
}

// --------------------------------------------------- groupnorm stats (8+8 groups)
__global__ __launch_bounds__(256) void k_gnstats(const float* __restrict__ spay,
                                                 const float* __restrict__ spey,
                                                 float* __restrict__ stats){
  int which = blockIdx.x >> 3;
  int b = (blockIdx.x >> 2) & 1;
  int g = blockIdx.x & 3;
  const float* src = which ? spey : spay;
  double sum = 0.0, sq = 0.0;
  for (int i = threadIdx.x; i < HW * 16; i += 256){
    int hw = i >> 4, cl = i & 15;
    float v = src[(size_t)((b << 14) + hw) * 64 + (g << 4) + cl];
    sum += v; sq += (double)v * v;
  }
  __shared__ double ls[256], lq[256];
  ls[threadIdx.x] = sum; lq[threadIdx.x] = sq;
  __syncthreads();
  for (int off = 128; off; off >>= 1){
    if (threadIdx.x < off){ ls[threadIdx.x] += ls[threadIdx.x + off]; lq[threadIdx.x] += lq[threadIdx.x + off]; }
    __syncthreads();
  }
  if (threadIdx.x == 0){
    double n = (double)(HW * 16);
    double mean = ls[0] / n;
    double var = lq[0] / n - mean * mean;
    stats[blockIdx.x * 2]     = (float)mean;
    stats[blockIdx.x * 2 + 1] = rsqrtf((float)var + 1e-5f);
  }
}

// ---------------------------------------------------------------- final fuse
__global__ __launch_bounds__(256) void k_fuse(const float* __restrict__ x,
                                              const float* __restrict__ spay,
                                              const float* __restrict__ spey,
                                              const float* __restrict__ stats,
                                              const float* __restrict__ gwa,
                                              const float* __restrict__ gba,
                                              const float* __restrict__ gwe,
                                              const float* __restrict__ gbe,
                                              const float* __restrict__ fw,
                                              float* __restrict__ out){
  int gid = blockIdx.x * 256 + threadIdx.x; // (b,c,hw) channel-major
  int hw = gid & (HW - 1);
  int bc = gid >> 14;
  int c = bc & 63, b = bc >> 6;
  int g = c >> 4;
  size_t tok = (size_t)((b << 14) + hw) * 64 + c;
  float xv = x[gid];
  int sid = ((b << 2) + g) * 2;
  float va = (spay[tok] - stats[sid]) * stats[sid + 1] * gwa[c] + gba[c];
  va = siluf(va) + xv;
  float ve = (spey[tok] - stats[16 + sid]) * stats[16 + sid + 1] * gwe[c] + gbe[c];
  ve = siluf(ve) + xv;
  float e0 = expf(fw[0]), e1 = expf(fw[1]);
  float w0 = e0 / (e0 + e1);
  out[gid] = va * w0 + ve * (1.f - w0) + xv;
}

extern "C" void kernel_launch(void* const* d_in, const int* in_sizes, int n_in,
                              void* d_out, int out_size, void* d_ws, size_t ws_size,
                              hipStream_t stream){
  (void)in_sizes; (void)n_in; (void)out_size; (void)ws_size;
  const float* x        = (const float*)d_in[0];
  const float* spa_in_w = (const float*)d_in[1];
  const float* spa_cw   = (const float*)d_in[2];
  const float* spa_cb   = (const float*)d_in[3];
  const float* spa_xpw  = (const float*)d_in[4];
  const float* spa_dtw  = (const float*)d_in[5];
  const float* spa_dtb  = (const float*)d_in[6];
  const float* spa_Al   = (const float*)d_in[7];
  const float* spa_D    = (const float*)d_in[8];
  const float* spa_ow   = (const float*)d_in[9];
  const float* spe_in_w = (const float*)d_in[10];
  const float* spe_cw   = (const float*)d_in[11];
  const float* spe_cb   = (const float*)d_in[12];
  const float* spe_xpw  = (const float*)d_in[13];
  const float* spe_dtw  = (const float*)d_in[14];
  const float* spe_dtb  = (const float*)d_in[15];
  const float* spe_Al   = (const float*)d_in[16];
  const float* spe_D    = (const float*)d_in[17];
  const float* spe_ow   = (const float*)d_in[18];
  const float* gwa      = (const float*)d_in[19];
  const float* gba      = (const float*)d_in[20];
  const float* gwe      = (const float*)d_in[21];
  const float* gbe      = (const float*)d_in[22];
  const float* fw       = (const float*)d_in[23];
  float* out = (float*)d_out;

  char* ws = (char*)d_ws;
  size_t off = 0;
  auto take = [&](size_t bytes) -> char* {
    char* p = ws + off;
    off += (bytes + 255) & ~(size_t)255;
    return p;
  };
  _Float16* xs_h  = (_Float16*)take((size_t)NT * 64 * 2);
  _Float16* inw_h = (_Float16*)take((size_t)256 * 64 * 2);
  _Float16* ow_h  = (_Float16*)take((size_t)64 * 128 * 2);
  float* xz    = (float*)take((size_t)NT * 256 * 4);
  float* xcv   = (float*)take((size_t)NT * 128 * 4);
  float* dbl   = (float*)take((size_t)NT * XPF * 4);
  float* delta = (float*)take((size_t)NT * 128 * 4);
  float* cP    = (float*)take((size_t)256 * 16 * NCHUNK * 4);
  float* cS    = (float*)take((size_t)256 * 16 * NCHUNK * 4);
  float* cH    = (float*)take((size_t)256 * 16 * NCHUNK * 4);
  float* ysc   = (float*)take((size_t)NT * 128 * 4);
  _Float16* y2h = (_Float16*)take((size_t)NT * 128 * 2);
  float* spay  = (float*)take((size_t)NT * 64 * 4);
  float* spey  = (float*)take((size_t)NT * 64 * 4);
  float* stats = (float*)take(32 * 4);

  k_wconv<<<64, 256, 0, stream>>>(spa_in_w, spa_ow, inw_h, ow_h);
  k_transpose<<<dim3(HW / 32, 2, Bn), dim3(32, 8), 0, stream>>>(x, xs_h);
  // in_proj: M=NT, N=256, K=64 ; 8 waves/block, 2 tiles/wave
  k_gemm_wmma<256, 64, 2><<<NT / 16, 256, 0, stream>>>(xs_h, inw_h, xz);
  k_conv<<<NT * 128 / 256, 256, 0, stream>>>(xz, spa_cw, spa_cb, xcv);
  k_xproj<<<NT * XPF / 256, 256, 0, stream>>>(xcv, spa_xpw, dbl);
  k_delta<<<NT * 128 / 256, 256, 0, stream>>>(dbl, spa_dtw, spa_dtb, delta);
  k_scan1<<<(256 * 16 * NCHUNK) / 256, 256, 0, stream>>>(delta, xcv, dbl, spa_Al, cP, cS);
  k_scan2<<<16, 256, 0, stream>>>(cP, cS, cH);
  k_scan3<<<(256 * 16 * NCHUNK) / 256, 256, 0, stream>>>(delta, xcv, dbl, spa_Al, cH, ysc);
  k_ymul<<<NT * 128 / 256, 256, 0, stream>>>(ysc, xcv, xz, spa_D, y2h);
  // out_proj: M=NT, N=64, K=128 ; 2 waves/block, 2 tiles/wave
  k_gemm_wmma<64, 128, 2><<<NT / 16, 64, 0, stream>>>(y2h, ow_h, spay);
  k_spe<<<NT / 8, 256, 0, stream>>>(x, spe_in_w, spe_cw, spe_cb, spe_xpw, spe_dtw, spe_dtb,
                                    spe_Al, spe_D, spe_ow, spey);
  k_gnstats<<<16, 256, 0, stream>>>(spay, spey, stats);
  k_fuse<<<(Bn * Cc * HW) / 256, 256, 0, stream>>>(x, spay, spey, stats, gwa, gba, gwe, gbe, fw, out);
}
